// Head_66649302499613
// MI455X (gfx1250) — matrix operand order repaired and checked by reading, not compiled, not run on hardware
//
#include <hip/hip_runtime.h>

// Fused single-head causal attention for MI455X (gfx1250, wave32, WMMA).
// B=2048, T=128, C=256, H=64.  One workgroup per batch element; x, Q, K, V and
// softmax all live in LDS/registers; x is read from HBM exactly once.
// Weights are pre-converted to f16 (chunk-major) in d_ws by a tiny prep kernel,
// then double-buffered into LDS with gfx1250 async loads (ASYNCcnt).
// Softmax is done entirely in registers (no f32 score buffer in LDS).

#define B_ 2048
#define T_ 128
#define C_ 256
#define H_ 64

typedef _Float16 half4_t __attribute__((ext_vector_type(4)));
typedef _Float16 half8_t __attribute__((ext_vector_type(8)));
typedef _Float16 v16h    __attribute__((ext_vector_type(16)));
typedef float    v8f     __attribute__((ext_vector_type(8)));
typedef float    f32x4   __attribute__((ext_vector_type(4)));

// ---- LDS layout (bytes). Padded row strides break bank conflicts. ----
// [0, 67584)          xs  : f16 [128][264]   (phase 2 only)
// [67584, 102400)     Wtc : f16 2 x [3][64][40] double buffer (phase 2)
//                     wei : f16 [128][136]   (softmax probs)    ALIASED
// [102400, 120832)    qs  : f16 [128][72]
// [120832, 139264)    ks  : f16 [128][72]
// [139264, 156672)    vT  : f16 [64][136]    (V transposed)
#define XS_STRIDE 264
#define W_STRIDE  40
#define QK_STRIDE 72
#define VT_STRIDE 136
#define WEI_STRIDE 136
#define OFF_WTC 67584
#define WTC_BUF_BYTES 15360    // 3*64*40*2
#define OFF_Q   102400
#define OFF_K   120832
#define OFF_V   139264
#define SMEM_TOTAL 156672

// d_ws weight layout: wt[kc][p][n][kk] f16, kc=0..7 (K chunk of 32), p=k/q/v,
// n=0..63 (output col), kk=0..31. Chunk block = 12288 B contiguous. 96 KB total.

// A-fragment for v_wmma_f32_16x16x32_f16 (16-bit A 16x32), this code's contiguous k order, not the ISA table order:
// lane (m = laneid&15, g = laneid>>4) holds K = 16g+e, e=0..15
// -> two contiguous 16B LDS loads (matches B-fragment order).
__device__ __forceinline__ v16h load_a_frag(const _Float16* p) {
    half8_t lo = *(const half8_t*)(p);
    half8_t hi = *(const half8_t*)(p + 8);
    v16h r;
#pragma unroll
    for (int i = 0; i < 8; ++i) { r[i] = lo[i]; r[i + 8] = hi[i]; }
    return r;
}

// B-fragment (16-bit B 32x16): lane (n = laneid&15, g) holds K = 16g+e,
// e=0..15 of column n -> one contiguous 32B run when stored [n][k].
__device__ __forceinline__ v16h load_b_frag(const _Float16* p) {
    half8_t lo = *(const half8_t*)(p);
    half8_t hi = *(const half8_t*)(p + 8);
    v16h r;
#pragma unroll
    for (int i = 0; i < 8; ++i) { r[i] = lo[i]; r[i + 8] = hi[i]; }
    return r;
}

__device__ __forceinline__ v8f wmma_f16(v16h a, v16h b, v8f c) {
    return __builtin_amdgcn_wmma_f32_16x16x32_f16(false, a, false, b,
                                                  (short)0, c, false, false);
}

// gfx1250 async global->LDS copy, tracked by ASYNCcnt (no VGPR data path).
__device__ __forceinline__ void async_ld_b128(unsigned lds_addr, unsigned voff,
                                              const _Float16* saddr) {
    asm volatile("global_load_async_to_lds_b128 %0, %1, %2"
                 :: "v"(lds_addr), "v"(voff), "s"(saddr) : "memory");
}

// ---------------- prep: W (f32 [256][64] x3) -> wt (f16 chunk-major) --------
__global__ __launch_bounds__(256)
void prep_weights_kernel(const float* __restrict__ Wk,
                         const float* __restrict__ Wq,
                         const float* __restrict__ Wv,
                         _Float16* __restrict__ wt) {
    int gid = blockIdx.x * 256 + threadIdx.x;    // [0, 3*256*64)
    int p = gid >> 14;                           // which weight matrix
    int r = gid & 16383;
    int c = r >> 6;                              // input row (C dim)
    int h = r & 63;                              // output col (H dim)
    const float* wp = (p == 0) ? Wk : ((p == 1) ? Wq : Wv);
    float v = wp[c * 64 + h];                    // coalesced over h
    int kc = c >> 5, kk = c & 31;
    wt[(((kc * 3 + p) * 64 + h) << 5) + kk] = (_Float16)v;
}

// ---------------- main fused attention kernel -------------------------------
__global__ __launch_bounds__(256)
void attn_head_kernel(const float* __restrict__ x,
                      const _Float16* __restrict__ wt,
                      float* __restrict__ out) {
    __shared__ __align__(16) unsigned char smem[SMEM_TOTAL];
    _Float16* xs  = (_Float16*)smem;
    _Float16* wei = (_Float16*)(smem + OFF_WTC);   // aliases Wtc (later phase)
    _Float16* qs  = (_Float16*)(smem + OFF_Q);
    _Float16* ks  = (_Float16*)(smem + OFF_K);
    _Float16* vT  = (_Float16*)(smem + OFF_V);

    const int b    = blockIdx.x;
    const int tidx = threadIdx.x;
    const int wave = tidx >> 5;
    const int lane = tidx & 31;
    const int g    = (lane >> 4) & 1;
    const int ln   = lane & 15;

    const float* xb = x + (size_t)b * (T_ * C_);

    // Wave-relative LDS base (generic-pointer low 32 bits = LDS byte offset).
    const unsigned lds_base = (unsigned)(uintptr_t)(&smem[0]);

    // Issue one weight chunk (12288 B) into LDS buffer `buf` asynchronously:
    // 3 x b128 per thread, per-lane LDS destinations (keeps the [n][40] pad).
    auto issue_chunk = [&](int kc, int buf) {
        unsigned dst_base = lds_base + OFF_WTC + (unsigned)(buf * WTC_BUF_BYTES);
#pragma unroll
        for (int i = 0; i < 3; ++i) {
            int idx8 = (tidx + i * 256) * 8;     // half index within chunk
            int p    = idx8 >> 11;
            int n    = (idx8 & 2047) >> 5;
            int kk0  = idx8 & 31;                // multiple of 8
            unsigned lds_addr = dst_base +
                (unsigned)((((p * 64 + n) * W_STRIDE) + kk0) * 2);
            unsigned voff = (unsigned)((kc * 6144 + idx8) * 2);
            async_ld_b128(lds_addr, voff, wt);
        }
    };

    // ---------- Phase 1: stage x -> LDS as f16 (vectorized float4) ----------
#pragma unroll
    for (int i = 0; i < (T_ * C_) / (256 * 4); ++i) {
        int elem = (tidx + i * 256) * 4;
        int row  = elem >> 8;
        int col  = elem & 255;
        f32x4 v = *(const f32x4*)(xb + elem);
        half4_t h;
        h[0] = (_Float16)v[0]; h[1] = (_Float16)v[1];
        h[2] = (_Float16)v[2]; h[3] = (_Float16)v[3];
        *(half4_t*)(xs + row * XS_STRIDE + col) = h;
    }

    // ---------- Phase 2: Q,K,V projections, async double-buffered weights ---
    // 96 tiles (3 proj x 8 Mtiles x 4 Ntiles); wave w owns tid = w + 8*i.
    // K loop fully unrolled: all LDS addresses fold into ds offset immediates.
    v8f acc[12] = {};
    issue_chunk(0, 0);
#pragma unroll
    for (int kc = 0; kc < 8; ++kc) {
        if (kc < 7) {
            issue_chunk(kc + 1, (kc + 1) & 1);
            asm volatile("s_wait_asynccnt 0x3" ::: "memory"); // chunk kc done
        } else {
            asm volatile("s_wait_asynccnt 0x0" ::: "memory");
        }
        __syncthreads();   // all waves' async portions + (iter 0) xs visible
        const _Float16* Wb = (const _Float16*)(smem + OFF_WTC +
                                               (kc & 1) * WTC_BUF_BYTES);
#pragma unroll
        for (int i = 0; i < 12; ++i) {
            int tid = wave + 8 * i;
            int p   = tid >> 5;
            int rem = tid & 31;
            int MT  = rem >> 2;
            int NT  = rem & 3;
            v16h a  = load_a_frag(xs + (MT * 16 + ln) * XS_STRIDE + kc * 32 + 16 * g);
            v16h bb = load_b_frag(Wb + (p * 64 + NT * 16 + ln) * W_STRIDE + 16 * g);
            acc[i] = wmma_f16(a, bb, acc[i]);
        }
        __syncthreads();   // buffer (kc&1) free before it is overwritten
    }
    // Write tiles: K,Q row-major; V transposed (so PV B-frags are contiguous).
#pragma unroll
    for (int i = 0; i < 12; ++i) {
        int tid = wave + 8 * i;
        int p   = tid >> 5;
        int rem = tid & 31;
        int MT  = rem >> 2;
        int NT  = rem & 3;
#pragma unroll
        for (int r = 0; r < 8; ++r) {
            int m = MT * 16 + r + 8 * g;     // C/D layout: VGPR r -> M = r + 8g
            int n = NT * 16 + ln;
            _Float16 hv = (_Float16)acc[i][r];
            if (p == 0)      ks[m * QK_STRIDE + n] = hv;
            else if (p == 1) qs[m * QK_STRIDE + n] = hv;
            else             vT[n * VT_STRIDE + m] = hv;
        }
    }
    __syncthreads();   // Q/K/V visible to all waves

    // ---------- Phase 3: scores + causal softmax, entirely in registers -----
    // Wave w owns score rows [16w, 16w+16): tiles MT=w, NT=0..7.
    // Lane (ln, g) holds, for each r, row m = 16w + r + 8g at columns 16i+ln.
    {
        const float scale = 0.0625f;  // 256^-0.5 (reference scales by C^-0.5)
        v8f sacc[8] = {};
#pragma unroll
        for (int c = 0; c < 2; ++c) {   // K-dim = H = 64 -> 2 chunks of 32
            v16h a = load_a_frag(qs + (wave * 16 + ln) * QK_STRIDE + c * 32 + 16 * g);
#pragma unroll
            for (int i = 0; i < 8; ++i) {
                v16h bb = load_b_frag(ks + (i * 16 + ln) * QK_STRIDE + c * 32 + 16 * g);
                sacc[i] = wmma_f16(a, bb, sacc[i]);
            }
        }
#pragma unroll
        for (int r = 0; r < 8; ++r) {
            int m = wave * 16 + r + 8 * g;
            float sv[8];
#pragma unroll
            for (int i = 0; i < 8; ++i) {
                int n = i * 16 + ln;
                sv[i] = (n <= m) ? sacc[i][r] * scale : -__builtin_inff();
            }
            float mx = sv[0];
#pragma unroll
            for (int i = 1; i < 8; ++i) mx = fmaxf(mx, sv[i]);
#pragma unroll
            for (int o = 8; o >= 1; o >>= 1)   // reduce within 16-lane half
                mx = fmaxf(mx, __shfl_xor(mx, o, 32));
            float sum = 0.0f;
#pragma unroll
            for (int i = 0; i < 8; ++i) { sv[i] = __expf(sv[i] - mx); sum += sv[i]; }
#pragma unroll
            for (int o = 8; o >= 1; o >>= 1)
                sum += __shfl_xor(sum, o, 32);
            float inv = 1.0f / sum;
#pragma unroll
            for (int i = 0; i < 8; ++i)
                wei[m * WEI_STRIDE + i * 16 + ln] = (_Float16)(sv[i] * inv);
        }
    }
    __syncthreads();   // defensive: wei region aliases the async buffers

    // ---------- Phase 4: out = wei @ V  (K-dim = T = 128, 4 chunks) ---------
    {
        v8f oacc[4] = {};
#pragma unroll
        for (int c = 0; c < 4; ++c) {
            v16h a = load_a_frag(wei + (wave * 16 + ln) * WEI_STRIDE + c * 32 + 16 * g);
#pragma unroll
            for (int i = 0; i < 4; ++i) {
                v16h bb = load_b_frag(vT + (i * 16 + ln) * VT_STRIDE + c * 32 + 16 * g);
                oacc[i] = wmma_f16(a, bb, oacc[i]);
            }
        }
        float* ob = out + (size_t)b * (T_ * H_);
#pragma unroll
        for (int i = 0; i < 4; ++i) {
#pragma unroll
            for (int r = 0; r < 8; ++r) {
                int m = wave * 16 + r + 8 * g;
                int n = i * 16 + ln;
                ob[m * H_ + n] = oacc[i][r];  // lanes 0..15 coalesce per row
            }
        }
    }
}

extern "C" void kernel_launch(void* const* d_in, const int* in_sizes, int n_in,
                              void* d_out, int out_size, void* d_ws, size_t ws_size,
                              hipStream_t stream) {
    (void)in_sizes; (void)n_in; (void)out_size; (void)ws_size;
    const float* x  = (const float*)d_in[0];
    const float* Wk = (const float*)d_in[1];
    const float* Wq = (const float*)d_in[2];
    const float* Wv = (const float*)d_in[3];
    _Float16* wt = (_Float16*)d_ws;        // needs 98,304 B of workspace
    float* out = (float*)d_out;
    prep_weights_kernel<<<dim3(192), dim3(256), 0, stream>>>(Wk, Wq, Wv, wt);
    attn_head_kernel<<<dim3(B_), dim3(256), 0, stream>>>(x, wt, out);
}
